// FujiSparseMoeBlock_71159018160284
// MI455X (gfx1250) — matrix operand order repaired and checked by reading, not compiled, not run on hardware
//
#include <hip/hip_runtime.h>
#include <cstdint>
#include <cstddef>

// Problem dims (fixed by the reference)
#define T_TOK 4096
#define HD    2048
#define IMOE  1408
#define ISH   5632
#define NE    8

typedef __attribute__((ext_vector_type(16))) __bf16 v16bf;
typedef __attribute__((ext_vector_type(8)))  float  v8f;

union FragAB { uint4 q[2]; v16bf v; };

// Wait until at most `n` async (ASYNCcnt) ops remain outstanding for this wave.
#define WAIT_ASYNC(n) asm volatile("s_wait_asynccnt " #n ::: "memory")

__device__ __forceinline__ unsigned short f2bf(float f) {
  unsigned int u = __float_as_uint(f);
  u += 0x7fffu + ((u >> 16) & 1u);   // round-to-nearest-even
  return (unsigned short)(u >> 16);
}

__device__ __forceinline__ unsigned lds_off(const void* p) {
  // Generic LDS address: low 32 bits are the wave-relative LDS byte offset.
  return (unsigned)(size_t)p;
}

// ---------------------------------------------------------------------------
// f32 -> bf16 bulk convert (vectorized; n multiple of 4)
// ---------------------------------------------------------------------------
__global__ void convert_f32_bf16(const float* __restrict__ in,
                                 unsigned short* __restrict__ out, size_t n4) {
  size_t i = (size_t)blockIdx.x * blockDim.x + threadIdx.x;
  size_t stride = (size_t)gridDim.x * blockDim.x;
  for (; i < n4; i += stride) {
    float4 f = reinterpret_cast<const float4*>(in)[i];
    unsigned int lo = (unsigned)f2bf(f.x) | ((unsigned)f2bf(f.y) << 16);
    unsigned int hi = (unsigned)f2bf(f.z) | ((unsigned)f2bf(f.w) << 16);
    reinterpret_cast<uint2*>(out)[i] = make_uint2(lo, hi);
  }
}

// ---------------------------------------------------------------------------
// Router: logits = x @ rw^T, softmax, top-2, renorm -> combine[T,8]
// Also sgate[t] = sigmoid(x[t] . shared_gate_w). One block per token.
// ---------------------------------------------------------------------------
__global__ void __launch_bounds__(256)
router_kernel(const float* __restrict__ x, const float* __restrict__ rw,
              const float* __restrict__ sgw, float* __restrict__ combine,
              float* __restrict__ sgate) {
  const int t = blockIdx.x;
  const int tid = threadIdx.x;
  const float* xt = x + (size_t)t * HD;

  float acc[9];
#pragma unroll
  for (int j = 0; j < 9; j++) acc[j] = 0.f;

  for (int k = tid; k < HD; k += 256) {
    float xv = xt[k];
#pragma unroll
    for (int e = 0; e < NE; e++) acc[e] += xv * rw[e * HD + k];
    acc[8] += xv * sgw[k];
  }

  __shared__ float red[256];
  __shared__ float res[9];
  for (int j = 0; j < 9; j++) {
    red[tid] = acc[j];
    __syncthreads();
    for (int s = 128; s > 0; s >>= 1) {
      if (tid < s) red[tid] += red[tid + s];
      __syncthreads();
    }
    if (tid == 0) res[j] = red[0];
    __syncthreads();
  }

  if (tid == 0) {
    float mx = res[0];
#pragma unroll
    for (int e = 1; e < NE; e++) mx = fmaxf(mx, res[e]);
    float p[NE], se = 0.f;
#pragma unroll
    for (int e = 0; e < NE; e++) { p[e] = expf(res[e] - mx); se += p[e]; }
#pragma unroll
    for (int e = 0; e < NE; e++) p[e] /= se;
    int i1 = 0;
#pragma unroll
    for (int e = 1; e < NE; e++) if (p[e] > p[i1]) i1 = e;
    int i2 = (i1 == 0) ? 1 : 0;
#pragma unroll
    for (int e = 0; e < NE; e++) if (e != i1 && p[e] > p[i2]) i2 = e;
    float s2 = p[i1] + p[i2];
#pragma unroll
    for (int e = 0; e < NE; e++) combine[(size_t)t * NE + e] = 0.f;
    combine[(size_t)t * NE + i1] = p[i1] / s2;
    combine[(size_t)t * NE + i2] = p[i2] / s2;
    sgate[t] = 1.f / (1.f + expf(-res[8]));
  }
}

// ---------------------------------------------------------------------------
// Async tile fetch: each thread owns a 32-byte chunk of the 64x32 bf16 tile.
// INST_OFFSET applies to BOTH LDS and global addresses, so chunk = 2x b128.
// ---------------------------------------------------------------------------
__device__ __forceinline__ void async_tile3(unsigned la, unsigned lg, unsigned lu,
                                            unsigned long long pa,
                                            unsigned long long pg,
                                            unsigned long long pu) {
  asm volatile(
      "global_load_async_to_lds_b128 %0, %3, off\n\t"
      "global_load_async_to_lds_b128 %0, %3, off offset:16\n\t"
      "global_load_async_to_lds_b128 %1, %4, off\n\t"
      "global_load_async_to_lds_b128 %1, %4, off offset:16\n\t"
      "global_load_async_to_lds_b128 %2, %5, off\n\t"
      "global_load_async_to_lds_b128 %2, %5, off offset:16"
      :: "v"(la), "v"(lg), "v"(lu), "v"(pa), "v"(pg), "v"(pu)
      : "memory");
}
__device__ __forceinline__ void async_tile2(unsigned la, unsigned lb,
                                            unsigned long long pa,
                                            unsigned long long pb) {
  asm volatile(
      "global_load_async_to_lds_b128 %0, %2, off\n\t"
      "global_load_async_to_lds_b128 %0, %2, off offset:16\n\t"
      "global_load_async_to_lds_b128 %1, %3, off\n\t"
      "global_load_async_to_lds_b128 %1, %3, off offset:16"
      :: "v"(la), "v"(lb), "v"(pa), "v"(pb)
      : "memory");
}

// ---------------------------------------------------------------------------
// LDS fragment helpers (tiles row-major [row][32] bf16).
// A fragment (16x32): lane m holds K {0..7,16..23} of row m; lane m+16 holds
// K {8..15,24..31}. B fragment (32x16): lane n holds K {0..7,16..23} of column n;
// lane n+16 holds K {8..15,24..31}.
// ---------------------------------------------------------------------------
__device__ __forceinline__ v16bf load_fragA(const unsigned short* s, int row, int lane) {
  FragAB f;
  const int c0 = (lane >> 4) * 8;
  const unsigned short* p = s + (size_t)row * 32 + c0;
  f.q[0] = *reinterpret_cast<const uint4*>(p);
  f.q[1] = *reinterpret_cast<const uint4*>(p + 16);
  return f.v;
}
__device__ __forceinline__ v16bf load_fragB(const unsigned short* s, int col, int lane) {
  FragAB f;
  const int k0 = (lane >> 4) * 8;
  const unsigned short* p = s + (size_t)col * 32 + k0;
  f.q[0] = *reinterpret_cast<const uint4*>(p);
  f.q[1] = *reinterpret_cast<const uint4*>(p + 16);
  return f.v;
}

// ---------------------------------------------------------------------------
// Fused gate/up GEMM: OUT[M,N] = bf16(silu(A@Bg^T) * (A@Bu^T)).
// A:[M,K], B*:[N,K] bf16 row-major. Tile 64x64x32, 128 thr (4 waves), each
// wave a 32x32 quadrant (2x2 WMMA). Double-buffered async LDS pipeline.
// ---------------------------------------------------------------------------
__global__ void __launch_bounds__(128)
gemm_gateup(const unsigned short* __restrict__ A,
            const unsigned short* __restrict__ Bg,
            const unsigned short* __restrict__ Bu,
            unsigned short* __restrict__ OUT, int M, int N, int K) {
  __shared__ __align__(16) unsigned short sA[2][64 * 32];
  __shared__ __align__(16) unsigned short sG[2][64 * 32];
  __shared__ __align__(16) unsigned short sU[2][64 * 32];

  const int tid = threadIdx.x;
  const int wave = tid >> 5, lane = tid & 31;
  const int m0 = blockIdx.x * 64, n0 = blockIdx.y * 64;
  const int wm = (wave & 1) * 32, wn = (wave >> 1) * 32;

  v8f zero = {0.f, 0.f, 0.f, 0.f, 0.f, 0.f, 0.f, 0.f};
  v8f accG[2][2], accU[2][2];
#pragma unroll
  for (int i = 0; i < 2; i++)
#pragma unroll
    for (int j = 0; j < 2; j++) { accG[i][j] = zero; accU[i][j] = zero; }

  const int lrow = tid >> 1;           // 0..63
  const int lcol = (tid & 1) * 16;     // 0 or 16
  const int tileoff = lrow * 32 + lcol;
  unsigned long long gA = (unsigned long long)(A + (size_t)(m0 + lrow) * K + lcol);
  unsigned long long gG = (unsigned long long)(Bg + (size_t)(n0 + lrow) * K + lcol);
  unsigned long long gU = (unsigned long long)(Bu + (size_t)(n0 + lrow) * K + lcol);
  unsigned lA[2] = {lds_off(&sA[0][tileoff]), lds_off(&sA[1][tileoff])};
  unsigned lG[2] = {lds_off(&sG[0][tileoff]), lds_off(&sG[1][tileoff])};
  unsigned lU[2] = {lds_off(&sU[0][tileoff]), lds_off(&sU[1][tileoff])};

  const int nk = K >> 5;
  async_tile3(lA[0], lG[0], lU[0], gA, gG, gU);

  for (int i = 0; i < nk; i++) {
    const int buf = i & 1;
    if (i + 1 < nk) {
      __syncthreads();  // everyone finished reading buf^1 (iteration i-1)
      unsigned long long adv = (unsigned long long)(i + 1) << 6;  // 32 elems * 2B
      async_tile3(lA[buf ^ 1], lG[buf ^ 1], lU[buf ^ 1], gA + adv, gG + adv, gU + adv);
      WAIT_ASYNC(6);    // tile i landed (only the 6 just-issued may remain)
    } else {
      WAIT_ASYNC(0);
    }
    __syncthreads();    // all waves' tile-i data visible in LDS

    v16bf af[2], gf[2], uf[2];
#pragma unroll
    for (int mi = 0; mi < 2; mi++)
      af[mi] = load_fragA(sA[buf], wm + mi * 16 + (lane & 15), lane);
#pragma unroll
    for (int ni = 0; ni < 2; ni++) {
      gf[ni] = load_fragB(sG[buf], wn + ni * 16 + (lane & 15), lane);
      uf[ni] = load_fragB(sU[buf], wn + ni * 16 + (lane & 15), lane);
    }
#pragma unroll
    for (int mi = 0; mi < 2; mi++)
#pragma unroll
      for (int ni = 0; ni < 2; ni++) {
        accG[mi][ni] = __builtin_amdgcn_wmma_f32_16x16x32_bf16(
            false, af[mi], false, gf[ni], (short)0, accG[mi][ni], false, false);
        accU[mi][ni] = __builtin_amdgcn_wmma_f32_16x16x32_bf16(
            false, af[mi], false, uf[ni], (short)0, accU[mi][ni], false, false);
      }
  }

  // epilogue: silu(g)*u -> bf16. C layout: vgpr r, lane L -> M=r+8*(L>=16), N=L&15
  const int colB = lane & 15;
  const int rsel = (lane >> 4) * 8;
#pragma unroll
  for (int mi = 0; mi < 2; mi++)
#pragma unroll
    for (int ni = 0; ni < 2; ni++)
#pragma unroll
      for (int r = 0; r < 8; r++) {
        int row = m0 + wm + mi * 16 + rsel + r;
        int col = n0 + wn + ni * 16 + colB;
        float g = accG[mi][ni][r];
        float u = accU[mi][ni][r];
        float s = g / (1.f + __expf(-g));
        OUT[(size_t)row * N + col] = f2bf(s * u);
      }
}

// ---------------------------------------------------------------------------
// Down GEMM with per-row scale: OUT[M,N] (op)= rowW[row*wstride] * (A @ B^T)
// ---------------------------------------------------------------------------
__global__ void __launch_bounds__(128)
gemm_down(const unsigned short* __restrict__ A,
          const unsigned short* __restrict__ B, float* __restrict__ OUT,
          const float* __restrict__ rowW, int wstride, int accumulate,
          int M, int N, int K) {
  __shared__ __align__(16) unsigned short sA[2][64 * 32];
  __shared__ __align__(16) unsigned short sB[2][64 * 32];

  const int tid = threadIdx.x;
  const int wave = tid >> 5, lane = tid & 31;
  const int m0 = blockIdx.x * 64, n0 = blockIdx.y * 64;
  const int wm = (wave & 1) * 32, wn = (wave >> 1) * 32;

  v8f zero = {0.f, 0.f, 0.f, 0.f, 0.f, 0.f, 0.f, 0.f};
  v8f acc[2][2];
#pragma unroll
  for (int i = 0; i < 2; i++)
#pragma unroll
    for (int j = 0; j < 2; j++) acc[i][j] = zero;

  const int lrow = tid >> 1;
  const int lcol = (tid & 1) * 16;
  const int tileoff = lrow * 32 + lcol;
  unsigned long long gA = (unsigned long long)(A + (size_t)(m0 + lrow) * K + lcol);
  unsigned long long gB = (unsigned long long)(B + (size_t)(n0 + lrow) * K + lcol);
  unsigned lA[2] = {lds_off(&sA[0][tileoff]), lds_off(&sA[1][tileoff])};
  unsigned lB[2] = {lds_off(&sB[0][tileoff]), lds_off(&sB[1][tileoff])};

  const int nk = K >> 5;
  async_tile2(lA[0], lB[0], gA, gB);

  for (int i = 0; i < nk; i++) {
    const int buf = i & 1;
    if (i + 1 < nk) {
      __syncthreads();
      unsigned long long adv = (unsigned long long)(i + 1) << 6;
      async_tile2(lA[buf ^ 1], lB[buf ^ 1], gA + adv, gB + adv);
      WAIT_ASYNC(4);
    } else {
      WAIT_ASYNC(0);
    }
    __syncthreads();

    v16bf af[2], bf[2];
#pragma unroll
    for (int mi = 0; mi < 2; mi++)
      af[mi] = load_fragA(sA[buf], wm + mi * 16 + (lane & 15), lane);
#pragma unroll
    for (int ni = 0; ni < 2; ni++)
      bf[ni] = load_fragB(sB[buf], wn + ni * 16 + (lane & 15), lane);
#pragma unroll
    for (int mi = 0; mi < 2; mi++)
#pragma unroll
      for (int ni = 0; ni < 2; ni++)
        acc[mi][ni] = __builtin_amdgcn_wmma_f32_16x16x32_bf16(
            false, af[mi], false, bf[ni], (short)0, acc[mi][ni], false, false);
  }

  const int colB = lane & 15;
  const int rsel = (lane >> 4) * 8;
#pragma unroll
  for (int mi = 0; mi < 2; mi++)
#pragma unroll
    for (int ni = 0; ni < 2; ni++)
#pragma unroll
      for (int r = 0; r < 8; r++) {
        int row = m0 + wm + mi * 16 + rsel + r;
        int col = n0 + wn + ni * 16 + colB;
        float w = rowW[(size_t)row * wstride];
        float v = acc[mi][ni][r] * w;
        size_t o = (size_t)row * N + col;
        if (accumulate) OUT[o] += v; else OUT[o] = v;
      }
}

// ---------------------------------------------------------------------------
// Host orchestration
// ---------------------------------------------------------------------------
extern "C" void kernel_launch(void* const* d_in, const int* in_sizes, int n_in,
                              void* d_out, int out_size, void* d_ws, size_t ws_size,
                              hipStream_t stream) {
  const float* x   = (const float*)d_in[0];  // [2,2048,2048]
  const float* rw  = (const float*)d_in[1];  // [8,2048]
  const float* gup = (const float*)d_in[2];  // [8,2816,2048]
  const float* dwn = (const float*)d_in[3];  // [8,2048,1408]
  const float* shg = (const float*)d_in[4];  // [5632,2048]
  const float* shu = (const float*)d_in[5];  // [5632,2048]
  const float* shd = (const float*)d_in[6];  // [2048,5632]
  const float* sgw = (const float*)d_in[7];  // [1,2048]
  float* out = (float*)d_out;                // [2,2048,2048]

  char* ws = (char*)d_ws;
  size_t off = 0;
  auto carve = [&](size_t bytes) -> char* {
    char* p = ws + off;
    off = (off + bytes + 255) & ~(size_t)255;
    return p;
  };
  unsigned short* xb    = (unsigned short*)carve((size_t)T_TOK * HD * 2);
  unsigned short* shg_b = (unsigned short*)carve((size_t)ISH * HD * 2);
  unsigned short* shu_b = (unsigned short*)carve((size_t)ISH * HD * 2);
  unsigned short* shd_b = (unsigned short*)carve((size_t)HD * ISH * 2);
  unsigned short* gup_b = (unsigned short*)carve((size_t)NE * 2 * IMOE * HD * 2);
  unsigned short* dwn_b = (unsigned short*)carve((size_t)NE * HD * IMOE * 2);
  unsigned short* act_s = (unsigned short*)carve((size_t)T_TOK * ISH * 2);
  unsigned short* act_m = (unsigned short*)carve((size_t)T_TOK * IMOE * 2);
  float* combine = (float*)carve((size_t)T_TOK * NE * 4);
  float* sgate   = (float*)carve((size_t)T_TOK * 4);
  (void)ws_size; (void)in_sizes; (void)n_in; (void)out_size;

  auto conv = [&](const float* src, unsigned short* dst, size_t n) {
    size_t n4 = n / 4;
    int blocks = (int)((n4 + 255) / 256);
    convert_f32_bf16<<<blocks, 256, 0, stream>>>(src, dst, n4);
  };
  conv(x,   xb,    (size_t)T_TOK * HD);
  conv(shg, shg_b, (size_t)ISH * HD);
  conv(shu, shu_b, (size_t)ISH * HD);
  conv(shd, shd_b, (size_t)HD * ISH);
  conv(gup, gup_b, (size_t)NE * 2 * IMOE * HD);
  conv(dwn, dwn_b, (size_t)NE * HD * IMOE);

  router_kernel<<<T_TOK, 256, 0, stream>>>(x, rw, sgw, combine, sgate);

  // Shared expert: act_s = silu(x@shg^T) * (x@shu^T); out = sgate * (act_s@shd^T)
  gemm_gateup<<<dim3(T_TOK / 64, ISH / 64), 128, 0, stream>>>(
      xb, shg_b, shu_b, act_s, T_TOK, ISH, HD);
  gemm_down<<<dim3(T_TOK / 64, HD / 64), 128, 0, stream>>>(
      act_s, shd_b, out, sgate, 1, /*accumulate=*/0, T_TOK, HD, ISH);

  // Experts (dense, weighted by combine; sequential on stream so += is safe)
  for (int e = 0; e < NE; e++) {
    const unsigned short* Bg = gup_b + (size_t)e * 2 * IMOE * HD;
    const unsigned short* Bu = Bg + (size_t)IMOE * HD;
    const unsigned short* Bd = dwn_b + (size_t)e * HD * IMOE;
    gemm_gateup<<<dim3(T_TOK / 64, IMOE / 64), 128, 0, stream>>>(
        xb, Bg, Bu, act_m, T_TOK, IMOE, HD);
    gemm_down<<<dim3(T_TOK / 64, HD / 64), 128, 0, stream>>>(
        act_m, Bd, out, combine + e, NE, /*accumulate=*/1, T_TOK, HD, IMOE);
  }
}
